// VNSimpleEncoder_47768626266580
// MI455X (gfx1250) — matrix operand order repaired and checked by reading, not compiled, not run on hardware
//
#include <hip/hip_runtime.h>
#include <hip/hip_bf16.h>

#define BATCH 2
#define NPTS  2048
#define KNN   20
#define EPSV  1e-6f
#define BNEPS 1e-5f

typedef __attribute__((ext_vector_type(16))) _Float16 v16h;
typedef __attribute__((ext_vector_type(8)))  _Float16 v8h;
typedef __attribute__((ext_vector_type(8)))  float    v8f;

// ---------------------------------------------------------------- utilities
__global__ void zero_kernel(float* p, int n) {
    int i = blockIdx.x * blockDim.x + threadIdx.x;
    for (; i < n; i += gridDim.x * blockDim.x) p[i] = 0.0f;
}

__device__ __forceinline__ int imin(int a, int b) { return a < b ? a : b; }

// ---------------------------------------------------------------- KNN (top-20 incl self)
__global__ __launch_bounds__(256) void knn_kernel(const float* __restrict__ x,
                                                  int* __restrict__ idx) {
    __shared__ float sx[256][3];
    int b = blockIdx.x >> 3;                       // 8 blocks per batch
    int n = (blockIdx.x & 7) * 256 + threadIdx.x;  // point id
    const float* xb = x + (long)b * NPTS * 3;
    float px = xb[n * 3 + 0], py = xb[n * 3 + 1], pz = xb[n * 3 + 2];
    float bestd[KNN];
    int   besti[KNN];
#pragma unroll
    for (int i = 0; i < KNN; ++i) { bestd[i] = 1e30f; besti[i] = 0; }
    for (int mt = 0; mt < NPTS; mt += 256) {
        int m = mt + threadIdx.x;
        sx[threadIdx.x][0] = xb[m * 3 + 0];
        sx[threadIdx.x][1] = xb[m * 3 + 1];
        sx[threadIdx.x][2] = xb[m * 3 + 2];
        __syncthreads();
        for (int j = 0; j < 256; ++j) {
            float dx = sx[j][0] - px, dy = sx[j][1] - py, dz = sx[j][2] - pz;
            float d = dx * dx + dy * dy + dz * dz;
            if (d < bestd[KNN - 1]) {
                int pos = KNN - 1;
                while (pos > 0 && bestd[pos - 1] > d) {
                    bestd[pos] = bestd[pos - 1];
                    besti[pos] = besti[pos - 1];
                    --pos;
                }
                bestd[pos] = d;
                besti[pos] = mt + j;
            }
        }
        __syncthreads();
    }
    long base = ((long)b * NPTS + n) * KNN;
#pragma unroll
    for (int i = 0; i < KNN; ++i) idx[base + i] = besti[i];
}

// ---------------------------------------------------------------- edge features (B,3,3,N,K)
__global__ __launch_bounds__(256) void feat_kernel(const float* __restrict__ x,
                                                   const int* __restrict__ idx,
                                                   float* __restrict__ feat) {
    long t = (long)blockIdx.x * 256 + threadIdx.x;
    if (t >= (long)BATCH * NPTS * KNN) return;
    int  k = (int)(t % KNN);
    long r = t / KNN;
    int  n = (int)(r % NPTS);
    int  b = (int)(r / NPTS);
    const float* xb = x + (long)b * NPTS * 3;
    float cx = xb[n * 3 + 0], cy = xb[n * 3 + 1], cz = xb[n * 3 + 2];
    int   m = idx[t];
    float nx = xb[m * 3 + 0], ny = xb[m * 3 + 1], nz = xb[m * 3 + 2];
    float ex = nx - cx, ey = ny - cy, ez = nz - cz;
    float crx = ny * cz - nz * cy;
    float cry = nz * cx - nx * cz;
    float crz = nx * cy - ny * cx;
    long S = (long)NPTS * KNN;
    long base = (long)b * 9 * S + (long)n * KNN + k;   // ((b*3+c)*3+d)*S + n*K + k
    feat[base + 0 * S] = ex;  feat[base + 1 * S] = ey;  feat[base + 2 * S] = ez;
    feat[base + 3 * S] = cx;  feat[base + 4 * S] = cy;  feat[base + 5 * S] = cz;
    feat[base + 6 * S] = crx; feat[base + 7 * S] = cry; feat[base + 8 * S] = crz;
}

// ---------------------------------------------------------------- VN linear + batch stats
// h: (B,Cin,3,S), W: (Cout,Cin) -> p: (B,Cout,3,S); stats[o]=sum||p||, stats[Cout+o]=sum||p||^2
__global__ __launch_bounds__(256) void lin_stats_kernel(const float* __restrict__ h,
                                                        const float* __restrict__ W,
                                                        float* __restrict__ p,
                                                        float* __restrict__ stats,
                                                        int Cin, int Cout, long S) {
    int  o    = blockIdx.y;
    long flat = (long)blockIdx.x * 256 + threadIdx.x;
    float nrm = 0.0f, nrm2 = 0.0f;
    if (flat < (long)BATCH * S) {
        int  b = (int)(flat / S);
        long s = flat % S;
        const float* hb = h + (long)b * Cin * 3 * S;
        float a0 = 0.f, a1 = 0.f, a2 = 0.f;
        for (int c = 0; c < Cin; ++c) {
            float w = W[o * Cin + c];
            const float* hc = hb + (long)c * 3 * S;
            a0 += w * hc[s];
            a1 += w * hc[S + s];
            a2 += w * hc[2 * S + s];
        }
        float* pb = p + ((long)b * Cout + o) * 3 * S;
        pb[s] = a0; pb[S + s] = a1; pb[2 * S + s] = a2;
        if (stats) {
            float nn = sqrtf(a0 * a0 + a1 * a1 + a2 * a2) + EPSV;
            nrm = nn; nrm2 = nn * nn;
        }
    }
    if (stats) {
        __shared__ float r1[256], r2[256];
        r1[threadIdx.x] = nrm; r2[threadIdx.x] = nrm2;
        __syncthreads();
        for (int st = 128; st > 0; st >>= 1) {
            if (threadIdx.x < st) {
                r1[threadIdx.x] += r1[threadIdx.x + st];
                r2[threadIdx.x] += r2[threadIdx.x + st];
            }
            __syncthreads();
        }
        if (threadIdx.x == 0) {
            atomicAdd(&stats[o], r1[0]);
            atomicAdd(&stats[Cout + o], r2[0]);
        }
    }
}

// ---------------------------------------------------------------- VN BN + leaky gate
__global__ __launch_bounds__(256) void leaky_kernel(const float* __restrict__ h,
                                                    const float* __restrict__ Wd,
                                                    const float* __restrict__ p,
                                                    const float* __restrict__ stats,
                                                    const float* __restrict__ gamma,
                                                    const float* __restrict__ beta,
                                                    float ns, int Cin, int Cout, long S,
                                                    float invCount, float* __restrict__ out) {
    int  o    = blockIdx.y;
    long flat = (long)blockIdx.x * 256 + threadIdx.x;
    if (flat >= (long)BATCH * S) return;
    int  b = (int)(flat / S);
    long s = flat % S;
    const float* hb = h + (long)b * Cin * 3 * S;
    float d0 = 0.f, d1 = 0.f, d2 = 0.f;
    for (int c = 0; c < Cin; ++c) {
        float w = Wd[o * Cin + c];
        const float* hc = hb + (long)c * 3 * S;
        d0 += w * hc[s];
        d1 += w * hc[S + s];
        d2 += w * hc[2 * S + s];
    }
    const float* pb = p + ((long)b * Cout + o) * 3 * S;
    float p0 = pb[s], p1 = pb[S + s], p2 = pb[2 * S + s];
    float nn  = sqrtf(p0 * p0 + p1 * p1 + p2 * p2) + EPSV;
    float mu  = stats[o] * invCount;
    float var = stats[Cout + o] * invCount - mu * mu;
    float nbn = (nn - mu) * rsqrtf(var + BNEPS) * gamma[o] + beta[o];
    float sc  = nbn / nn;
    p0 *= sc; p1 *= sc; p2 *= sc;
    float dot = p0 * d0 + p1 * d1 + p2 * d2;
    float dsq = d0 * d0 + d1 * d1 + d2 * d2 + EPSV;
    float f   = dot / dsq;
    float q0 = (dot >= 0.f) ? p0 : (p0 - f * d0);
    float q1 = (dot >= 0.f) ? p1 : (p1 - f * d1);
    float q2 = (dot >= 0.f) ? p2 : (p2 - f * d2);
    float* ob = out + ((long)b * Cout + o) * 3 * S;
    ob[s]         = ns * p0 + (1.f - ns) * q0;
    ob[S + s]     = ns * p1 + (1.f - ns) * q1;
    ob[2 * S + s] = ns * p2 + (1.f - ns) * q2;
}

// ---------------------------------------------------------------- layer-1 gate + mean over k
__global__ __launch_bounds__(256) void leaky_pool_kernel(const float* __restrict__ feat,
                                                         const float* __restrict__ Wd,
                                                         const float* __restrict__ p,
                                                         const float* __restrict__ stats,
                                                         const float* __restrict__ gamma,
                                                         const float* __restrict__ beta,
                                                         int Cout, float invCount,
                                                         float* __restrict__ out) {
    int  o    = blockIdx.y;
    long flat = (long)blockIdx.x * 256 + threadIdx.x;
    if (flat >= (long)BATCH * NPTS) return;
    int b = (int)(flat / NPTS);
    int n = (int)(flat % NPTS);
    long S = (long)NPTS * KNN;
    float mu  = stats[o] * invCount;
    float var = stats[Cout + o] * invCount - mu * mu;
    float gsc = rsqrtf(var + BNEPS) * gamma[o];
    float be  = beta[o];
    float w0 = Wd[o * 3 + 0], w1 = Wd[o * 3 + 1], w2 = Wd[o * 3 + 2];
    const float* fb = feat + (long)b * 9 * S;
    const float* pb = p + ((long)b * Cout + o) * 3 * S;
    float a0 = 0.f, a1 = 0.f, a2 = 0.f;
    for (int k = 0; k < KNN; ++k) {
        long s = (long)n * KNN + k;
        float d0 = w0 * fb[0 * S + s] + w1 * fb[3 * S + s] + w2 * fb[6 * S + s];
        float d1 = w0 * fb[1 * S + s] + w1 * fb[4 * S + s] + w2 * fb[7 * S + s];
        float d2 = w0 * fb[2 * S + s] + w1 * fb[5 * S + s] + w2 * fb[8 * S + s];
        float p0 = pb[s], p1 = pb[S + s], p2 = pb[2 * S + s];
        float nn  = sqrtf(p0 * p0 + p1 * p1 + p2 * p2) + EPSV;
        float nbn = (nn - mu) * gsc + be;
        float sc  = nbn / nn;
        p0 *= sc; p1 *= sc; p2 *= sc;
        float dot = p0 * d0 + p1 * d1 + p2 * d2;
        float dsq = d0 * d0 + d1 * d1 + d2 * d2 + EPSV;
        float f   = dot / dsq;
        a0 += (dot >= 0.f) ? p0 : (p0 - f * d0);   // ns = 0
        a1 += (dot >= 0.f) ? p1 : (p1 - f * d1);
        a2 += (dot >= 0.f) ? p2 : (p2 - f * d2);
    }
    float inv = 1.0f / KNN;
    float* ob = out + ((long)b * Cout + o) * 3 * NPTS;
    ob[n]            = a0 * inv;
    ob[NPTS + n]     = a1 * inv;
    ob[2 * NPTS + n] = a2 * inv;
}

// ---------------------------------------------------------------- VN BN only + z accumulation
__global__ __launch_bounds__(256) void bn_z_kernel(const float* __restrict__ p,
                                                   const float* __restrict__ stats,
                                                   const float* __restrict__ gamma,
                                                   const float* __restrict__ beta,
                                                   int Cout, float invCount,
                                                   float* __restrict__ out,
                                                   float* __restrict__ zacc) {
    int  o    = blockIdx.y;
    long flat = (long)blockIdx.x * 256 + threadIdx.x;
    float o0 = 0.f, o1 = 0.f, o2 = 0.f;
    if (flat < (long)BATCH * NPTS) {
        int b = (int)(flat / NPTS);
        int n = (int)(flat % NPTS);
        const float* pb = p + ((long)b * Cout + o) * 3 * NPTS;
        float p0 = pb[n], p1 = pb[NPTS + n], p2 = pb[2 * NPTS + n];
        float nn  = sqrtf(p0 * p0 + p1 * p1 + p2 * p2) + EPSV;
        float mu  = stats[o] * invCount;
        float var = stats[Cout + o] * invCount - mu * mu;
        float nbn = (nn - mu) * rsqrtf(var + BNEPS) * gamma[o] + beta[o];
        float sc  = nbn / nn;
        o0 = p0 * sc; o1 = p1 * sc; o2 = p2 * sc;
        float* ob = out + ((long)b * Cout + o) * 3 * NPTS;
        ob[n] = o0; ob[NPTS + n] = o1; ob[2 * NPTS + n] = o2;
    }
    __shared__ float r0[256], r1[256], r2[256];
    r0[threadIdx.x] = o0; r1[threadIdx.x] = o1; r2[threadIdx.x] = o2;
    __syncthreads();
    for (int st = 128; st > 0; st >>= 1) {
        if (threadIdx.x < st) {
            r0[threadIdx.x] += r0[threadIdx.x + st];
            r1[threadIdx.x] += r1[threadIdx.x + st];
            r2[threadIdx.x] += r2[threadIdx.x + st];
        }
        __syncthreads();
    }
    if (threadIdx.x == 0) {
        int bblk = (int)(((long)blockIdx.x * 256) / NPTS);
        float inv = 1.0f / NPTS;
        atomicAdd(&zacc[((long)bblk * Cout + o) * 3 + 0], r0[0] * inv);
        atomicAdd(&zacc[((long)bblk * Cout + o) * 3 + 1], r1[0] * inv);
        atomicAdd(&zacc[((long)bblk * Cout + o) * 3 + 2], r2[0] * inv);
    }
}

__global__ void write_z_kernel(const float* zacc, float* out) {
    int i = blockIdx.x * blockDim.x + threadIdx.x;
    if (i < BATCH * 170 * 3) out[i] = zacc[i];
}

// ---------------------------------------------------------------- concat h170 | broadcast z
__global__ __launch_bounds__(256) void h340_kernel(const float* __restrict__ h170,
                                                   const float* __restrict__ zacc,
                                                   float* __restrict__ h340) {
    long t = (long)blockIdx.x * 256 + threadIdx.x;
    if (t >= (long)BATCH * 340 * NPTS) return;
    int  n = (int)(t % NPTS);
    long r = t / NPTS;
    int  o = (int)(r % 340);
    int  b = (int)(r / 340);
    float v0, v1, v2;
    if (o < 170) {
        const float* hb = h170 + ((long)b * 170 + o) * 3 * NPTS;
        v0 = hb[n]; v1 = hb[NPTS + n]; v2 = hb[2 * NPTS + n];
    } else {
        const float* zb = zacc + ((long)b * 170 + (o - 170)) * 3;
        v0 = zb[0]; v1 = zb[1]; v2 = zb[2];
    }
    float* ob = h340 + ((long)b * 340 + o) * 3 * NPTS;
    ob[n] = v0; ob[NPTS + n] = v1; ob[2 * NPTS + n] = v2;
}

// ---------------------------------------------------------------- hs = h340 x z0 -> t (B,1020,N)
__global__ __launch_bounds__(256) void hs_t_kernel(const float* __restrict__ h340,
                                                   const float* __restrict__ z0,
                                                   float* __restrict__ t) {
    long id = (long)blockIdx.x * 256 + threadIdx.x;
    if (id >= (long)BATCH * 340 * NPTS) return;
    int  n = (int)(id % NPTS);
    long r = id / NPTS;
    int  i = (int)(r % 340);
    int  b = (int)(r / 340);
    const float* hb = h340 + ((long)b * 340 + i) * 3 * NPTS;
    float h0 = hb[n], h1 = hb[NPTS + n], h2 = hb[2 * NPTS + n];
    const float* zb = z0 + (long)b * 9 * NPTS;   // ((b*3+j)*3+kd)*N+n
#pragma unroll
    for (int kd = 0; kd < 3; ++kd) {
        float v = h0 * zb[(0 * 3 + kd) * NPTS + n] +
                  h1 * zb[(1 * 3 + kd) * NPTS + n] +
                  h2 * zb[(2 * 3 + kd) * NPTS + n];
        t[((long)b * 1020 + (long)i * 3 + kd) * NPTS + n] = v;
    }
}

// ---------------------------------------------------------------- WMMA GEMM (f16 in, f32 acc)
// C[m,n] = act( A[m,:] * B[:,n] + bias[m] ), A: MxK weights, B: KxN per-batch, C: MxN per-batch
#define TILE_K 32
__global__ __launch_bounds__(256) void gemm_wmma_kernel(const float* __restrict__ A,
                                                        const float* __restrict__ Bm,
                                                        const float* __restrict__ bias,
                                                        float* __restrict__ C,
                                                        int M, int Kd, int Ncols,
                                                        long bStride, long cStride, int relu) {
    __shared__ _Float16 As[64][40];    // [m][k], k contiguous, 80B row stride (16B aligned)
    __shared__ _Float16 Bs[128][40];   // [n][k], k contiguous (transposed stage)
    const float* Bb = Bm + (long)blockIdx.z * bStride;
    float*       Cb = C  + (long)blockIdx.z * cStride;
    int tid  = threadIdx.x;
    int lane = tid & 31;
    int wave = tid >> 5;
    int waveM = (wave >> 2) * 32;   // 0 / 32
    int waveN = (wave & 3) * 32;    // 0..96
    int blockM = blockIdx.y * 64;
    int blockN = blockIdx.x * 128;
    int lrow = lane & 15;
    int kg   = lane >> 4;

    // staging coordinates
    int ar = tid >> 2, ac0 = (tid & 3) * 8;           // A: row, 8 consecutive k
    int bk = tid >> 3, bn0 = (tid & 7) * 16;          // B: k row, 16 consecutive n
    // clamped (always-valid) row pointers; out-of-range M rows are never stored,
    // so A rows only need valid addresses, not masking.
    const float* Arow = A + (long)imin(blockM + ar, M - 1) * Kd;

    v8f acc[2][2] = {};
    int Kt = (Kd + TILE_K - 1) / TILE_K;
    for (int kt = 0; kt < Kt; ++kt) {
        int k0 = kt * TILE_K;
        if (kt + 1 < Kt) {  // speculative prefetch of next K-tile (global_prefetch_b8)
            __builtin_prefetch(&Bb[(long)(k0 + TILE_K + bk) * Ncols + blockN + bn0], 0, 1);
            __builtin_prefetch(&Arow[k0 + TILE_K + ac0], 0, 1);
        }
        if (k0 + TILE_K <= Kd) {
            // ---- fast path: full K-tile, unguarded vector loads
            {   // A: 2x float4 -> 8 halves -> one b128 LDS store
                float4 va = *(const float4*)(Arow + k0 + ac0);
                float4 vb = *(const float4*)(Arow + k0 + ac0 + 4);
                v8h hv;
                hv[0] = (_Float16)va.x; hv[1] = (_Float16)va.y;
                hv[2] = (_Float16)va.z; hv[3] = (_Float16)va.w;
                hv[4] = (_Float16)vb.x; hv[5] = (_Float16)vb.y;
                hv[6] = (_Float16)vb.z; hv[7] = (_Float16)vb.w;
                *(v8h*)&As[ar][ac0] = hv;
            }
            {   // B: 4x float4 (16 consecutive n), transpose-scatter to Bs[n][k]
                const float* Bp = Bb + (long)(k0 + bk) * Ncols + blockN + bn0;
                float4 v0 = *(const float4*)(Bp + 0);
                float4 v1 = *(const float4*)(Bp + 4);
                float4 v2 = *(const float4*)(Bp + 8);
                float4 v3 = *(const float4*)(Bp + 12);
                float tmp[16] = { v0.x, v0.y, v0.z, v0.w, v1.x, v1.y, v1.z, v1.w,
                                  v2.x, v2.y, v2.z, v2.w, v3.x, v3.y, v3.z, v3.w };
#pragma unroll
                for (int j = 0; j < 16; ++j)
                    Bs[bn0 + j][bk] = (_Float16)tmp[j];
            }
        } else {
            // ---- remainder tile: branchless clamped loads + zero-select
            {
                float tmp[8];
#pragma unroll
                for (int c = 0; c < 8; ++c) {
                    int gk = k0 + ac0 + c;
                    tmp[c] = Arow[imin(gk, Kd - 1)];
                }
                v8h hv;
#pragma unroll
                for (int c = 0; c < 8; ++c) {
                    int gk = k0 + ac0 + c;
                    hv[c] = (_Float16)((gk < Kd) ? tmp[c] : 0.0f);
                }
                *(v8h*)&As[ar][ac0] = hv;
            }
            {
                int gk = k0 + bk;
                const float* Bp = Bb + (long)imin(gk, Kd - 1) * Ncols + blockN + bn0;
                float4 v0 = *(const float4*)(Bp + 0);
                float4 v1 = *(const float4*)(Bp + 4);
                float4 v2 = *(const float4*)(Bp + 8);
                float4 v3 = *(const float4*)(Bp + 12);
                float msk = (gk < Kd) ? 1.0f : 0.0f;
                float tmp[16] = { v0.x, v0.y, v0.z, v0.w, v1.x, v1.y, v1.z, v1.w,
                                  v2.x, v2.y, v2.z, v2.w, v3.x, v3.y, v3.z, v3.w };
#pragma unroll
                for (int j = 0; j < 16; ++j)
                    Bs[bn0 + j][bk] = (_Float16)(tmp[j] * msk);
            }
        }
        __syncthreads();
        v16h afrag[2], bfrag[2];
#pragma unroll
        for (int tm = 0; tm < 2; ++tm) {
            int mr = waveM + tm * 16 + lrow;
            union { v16h v; unsigned u[8]; } ua;
#pragma unroll
            for (int p = 0; p < 8; ++p) {
                int kA = (p < 4) ? (kg * 8 + 2 * p) : (16 + kg * 8 + 2 * (p - 4));
                ua.u[p] = *(const unsigned*)&As[mr][kA];
            }
            afrag[tm] = ua.v;
            int nr = waveN + tm * 16 + lrow;
            union { v16h v; unsigned u[8]; } ub;
#pragma unroll
            for (int p = 0; p < 8; ++p) {
                int kB = (p < 4) ? (kg * 8 + 2 * p) : (16 + kg * 8 + 2 * (p - 4));
                ub.u[p] = *(const unsigned*)&Bs[nr][kB];
            }
            bfrag[tm] = ub.v;
        }
#pragma unroll
        for (int i = 0; i < 2; ++i)
#pragma unroll
            for (int j = 0; j < 2; ++j)
                acc[i][j] = __builtin_amdgcn_wmma_f32_16x16x32_f16(
                    false, afrag[i], false, bfrag[j], (short)0, acc[i][j], false, false);
        __syncthreads();
    }
#pragma unroll
    for (int i = 0; i < 2; ++i) {
#pragma unroll
        for (int j = 0; j < 2; ++j) {
#pragma unroll
            for (int r = 0; r < 8; ++r) {
                int m = blockM + waveM + i * 16 + kg * 8 + r;
                int n = blockN + waveN + j * 16 + lrow;
                if (m < M && n < Ncols) {
                    float v = acc[i][j][r];
                    if (bias) v += bias[m];
                    if (relu) v = fmaxf(v, 0.0f);
                    Cb[(long)m * Ncols + n] = v;
                }
            }
        }
    }
}

// ================================================================ host
extern "C" void kernel_launch(void* const* d_in, const int* in_sizes, int n_in,
                              void* d_out_v, int out_size, void* d_ws, size_t ws_size,
                              hipStream_t stream) {
    const float* x      = (const float*)d_in[0];
    const float* Wf_pos = (const float*)d_in[1];
    const float* Wd_pos = (const float*)d_in[2];
    const float* g_pos  = (const float*)d_in[3];
    const float* be_pos = (const float*)d_in[4];
    const float* Wf1    = (const float*)d_in[5];
    const float* Wd1    = (const float*)d_in[6];
    const float* g1     = (const float*)d_in[7];
    const float* be1    = (const float*)d_in[8];
    const float* Wf2    = (const float*)d_in[9];
    const float* Wd2    = (const float*)d_in[10];
    const float* g2     = (const float*)d_in[11];
    const float* be2    = (const float*)d_in[12];
    const float* W3     = (const float*)d_in[13];
    const float* g3     = (const float*)d_in[14];
    const float* be3    = (const float*)d_in[15];
    const float* Ws1f   = (const float*)d_in[16];
    const float* Ws1d   = (const float*)d_in[17];
    const float* gs1    = (const float*)d_in[18];
    const float* bs1    = (const float*)d_in[19];
    const float* Ws2f   = (const float*)d_in[20];
    const float* Ws2d   = (const float*)d_in[21];
    const float* gs2    = (const float*)d_in[22];
    const float* bs2    = (const float*)d_in[23];
    const float* Wstd   = (const float*)d_in[24];
    const float* Wt1    = (const float*)d_in[25];
    const float* Wt2    = (const float*)d_in[26];
    const float* Wt3    = (const float*)d_in[27];
    const float* bt3    = (const float*)d_in[28];
    float* out = (float*)d_out_v;

    const long SNK = (long)NPTS * KNN;   // 40960
    const long SN  = NPTS;

    float* ws = (float*)d_ws;
    long off = 0;
    int*   idxbuf = (int*)(ws + off); off += (long)BATCH * NPTS * KNN;   // 81920
    float* feat   = ws + off;         off += (long)BATCH * 9 * SNK;      // 737280
    float* pbuf   = ws + off;         off += (long)BATCH * 21 * 3 * SNK; // 5160960 (largest p)
    float* h1     = ws + off;         off += (long)BATCH * 21 * 3 * SN;
    float* h2     = ws + off;         off += (long)BATCH * 21 * 3 * SN;
    float* h3     = ws + off;         off += (long)BATCH * 42 * 3 * SN;
    float* h170   = ws + off;         off += (long)BATCH * 170 * 3 * SN;
    float* zacc   = ws + off;         off += 1024;
    float* h340   = ws + off;         off += (long)BATCH * 340 * 3 * SN;
    float* z0a    = ws + off;         off += (long)BATCH * 170 * 3 * SN;
    float* z0b    = ws + off;         off += (long)BATCH * 85 * 3 * SN;
    float* z0s    = ws + off;         off += (long)BATCH * 3 * 3 * SN;
    float* tb     = ws + off;         off += (long)BATCH * 1020 * SN;
    float* t1b    = ws + off;         off += (long)BATCH * 1020 * SN;
    float* t2b    = ws + off;         off += (long)BATCH * 1020 * SN;
    float* stats  = ws + off;         off += 1024;

    // ---- KNN + edge features
    knn_kernel<<<dim3(16), dim3(256), 0, stream>>>(x, idxbuf);
    feat_kernel<<<dim3((unsigned)((BATCH * SNK + 255) / 256)), dim3(256), 0, stream>>>(x, idxbuf, feat);

    int gNK = (int)((BATCH * SNK + 255) / 256);   // 320
    int gN  = (int)((BATCH * SN + 255) / 256);    // 16

    // ---- layer 1: feat -> 21 (ns=0, mean over k)
    zero_kernel<<<4, 256, 0, stream>>>(stats, 1024);
    lin_stats_kernel<<<dim3(gNK, 21), 256, 0, stream>>>(feat, Wf_pos, pbuf, stats, 3, 21, SNK);
    leaky_pool_kernel<<<dim3(gN, 21), 256, 0, stream>>>(feat, Wd_pos, pbuf, stats, g_pos, be_pos,
                                                        21, 1.0f / (BATCH * SNK), h1);
    // ---- layer 2: 21 -> 21 (ns=0)
    zero_kernel<<<4, 256, 0, stream>>>(stats, 1024);
    lin_stats_kernel<<<dim3(gN, 21), 256, 0, stream>>>(h1, Wf1, pbuf, stats, 21, 21, SN);
    leaky_kernel<<<dim3(gN, 21), 256, 0, stream>>>(h1, Wd1, pbuf, stats, g1, be1, 0.0f, 21, 21, SN,
                                                   1.0f / (BATCH * SN), h2);
    // ---- layer 3: 21 -> 42 (ns=0)
    zero_kernel<<<4, 256, 0, stream>>>(stats, 1024);
    lin_stats_kernel<<<dim3(gN, 42), 256, 0, stream>>>(h2, Wf2, pbuf, stats, 21, 42, SN);
    leaky_kernel<<<dim3(gN, 42), 256, 0, stream>>>(h2, Wd2, pbuf, stats, g2, be2, 0.0f, 21, 42, SN,
                                                   1.0f / (BATCH * SN), h3);
    // ---- layer 4: 42 -> 170 linear + BN; z = mean_n
    zero_kernel<<<4, 256, 0, stream>>>(stats, 1024);
    lin_stats_kernel<<<dim3(gN, 170), 256, 0, stream>>>(h3, W3, pbuf, stats, 42, 170, SN);
    zero_kernel<<<4, 256, 0, stream>>>(zacc, 1024);
    bn_z_kernel<<<dim3(gN, 170), 256, 0, stream>>>(pbuf, stats, g3, be3, 170,
                                                   1.0f / (BATCH * SN), h170, zacc);
    write_z_kernel<<<4, 256, 0, stream>>>(zacc, out);
    // ---- concat -> h340
    h340_kernel<<<dim3((unsigned)(((long)BATCH * 340 * SN + 255) / 256)), 256, 0, stream>>>(h170, zacc, h340);
    // ---- std branch: 340 -> 170 (ns=0.2)
    zero_kernel<<<4, 256, 0, stream>>>(stats, 1024);
    lin_stats_kernel<<<dim3(gN, 170), 256, 0, stream>>>(h340, Ws1f, pbuf, stats, 340, 170, SN);
    leaky_kernel<<<dim3(gN, 170), 256, 0, stream>>>(h340, Ws1d, pbuf, stats, gs1, bs1, 0.2f, 340, 170,
                                                    SN, 1.0f / (BATCH * SN), z0a);
    // ---- std branch: 170 -> 85 (ns=0.2)
    zero_kernel<<<4, 256, 0, stream>>>(stats, 1024);
    lin_stats_kernel<<<dim3(gN, 85), 256, 0, stream>>>(z0a, Ws2f, pbuf, stats, 170, 85, SN);
    leaky_kernel<<<dim3(gN, 85), 256, 0, stream>>>(z0a, Ws2d, pbuf, stats, gs2, bs2, 0.2f, 170, 85,
                                                   SN, 1.0f / (BATCH * SN), z0b);
    // ---- std branch: 85 -> 3 plain linear
    lin_stats_kernel<<<dim3(gN, 3), 256, 0, stream>>>(z0b, Wstd, z0s, nullptr, 85, 3, SN);
    // ---- invariant features -> t (B,1020,N)
    hs_t_kernel<<<dim3((unsigned)(((long)BATCH * 340 * SN + 255) / 256)), 256, 0, stream>>>(h340, z0s, tb);

    // ---- big WMMA GEMMs
    long sB = 1020L * NPTS;
    dim3 g12((NPTS + 127) / 128, (1020 + 63) / 64, BATCH);
    gemm_wmma_kernel<<<g12, 256, 0, stream>>>(Wt1, tb, nullptr, t1b, 1020, 1020, NPTS, sB, sB, 1);
    gemm_wmma_kernel<<<g12, 256, 0, stream>>>(Wt2, t1b, nullptr, t2b, 1020, 1020, NPTS, sB, sB, 1);
    dim3 g3d((NPTS + 127) / 128, (14450 + 63) / 64, BATCH);
    gemm_wmma_kernel<<<g3d, 256, 0, stream>>>(Wt3, t2b, bt3, out + 1020, 14450, 1020, NPTS,
                                              sB, 14450L * NPTS, 0);

    (void)in_sizes; (void)n_in; (void)out_size; (void)ws_size;
}
